// Attention_7559142441592
// MI455X (gfx1250) — matrix operand order repaired and checked by reading, not compiled, not run on hardware
//
#include <hip/hip_runtime.h>

typedef unsigned short u16;
typedef __attribute__((ext_vector_type(16))) __bf16 v16bf;
typedef __attribute__((ext_vector_type(8)))  float  v8f;
typedef __attribute__((ext_vector_type(4)))  unsigned int v4u;
typedef __attribute__((ext_vector_type(4)))  float  v4f;
typedef __attribute__((ext_vector_type(4)))  u16    v4h;

union BF16x16 { v16bf v; u16 u[16]; v4u q[2]; };

__device__ __forceinline__ u16 f2bf(float f){
  unsigned u = __float_as_uint(f);
  u += 0x7FFFu + ((u >> 16) & 1u);          // round-to-nearest-even
  return (u16)(u >> 16);
}
__device__ __forceinline__ float bf2f(u16 h){ return __uint_as_float(((unsigned)h) << 16); }

__device__ __forceinline__ v8f wmma_bf16(v16bf a, v16bf b, v8f c){
  return __builtin_amdgcn_wmma_f32_16x16x32_bf16(false, a, false, b, (short)0, c, false, false);
}

// A fragment (16x32 bf16, row-major src): lane holds row = lane&15.
// element e -> k = e + half*8 + (e>=8 ? 8 : 0)  => chunks at kbase+half*8 and kbase+16+half*8
__device__ __forceinline__ v16bf ldA(const u16* row, int kbase, int half){
  BF16x16 f;
  f.q[0] = *(const v4u*)(row + kbase + half*8);
  f.q[1] = *(const v4u*)(row + kbase + 16 + half*8);
  return f.v;
}
// B fragment (32x16 bf16): lane holds column n = lane&15; its 32 k-values are
// contiguous at `col`; element e -> k = e + half*8 + (e>=8 ? 8 : 0)  => chunks at kbase+half*8 and kbase+16+half*8
__device__ __forceinline__ v16bf ldB(const u16* col, int kbase, int half){
  BF16x16 f;
  const u16* p = col + kbase + half*8;
  f.q[0] = *(const v4u*)p;
  f.q[1] = *(const v4u*)(p + 16);
  return f.v;
}

// ---------------- pass 0: fp32 -> bf16 convert ----------------
__global__ void cvt_kernel(const float* __restrict__ in, u16* __restrict__ out, int n4){
  int i = blockIdx.x*blockDim.x + threadIdx.x;
  if (i >= n4) return;
  v4f v = *(const v4f*)(in + (size_t)i*4);
  v4h h; h[0]=f2bf(v[0]); h[1]=f2bf(v[1]); h[2]=f2bf(v[2]); h[3]=f2bf(v[3]);
  *(v4h*)(out + (size_t)i*4) = h;
}

// ---------------- pass 0b: zero pad tokens 1032..1055 of Q/K (row-major) and Vt (transposed) ----
__global__ void pad_kernel(u16* __restrict__ Qb, u16* __restrict__ Kb, u16* __restrict__ Vt){
  int i = blockIdx.x*blockDim.x + threadIdx.x;       // 49152
  if (i >= 49152) return;
  v4h z = (v4h)0;
  { // Qb/Kb: per b2 the pad is contiguous 1536 halfs = 384 v4h
    int b2 = i / 384, j = i - b2*384;
    size_t off = ((size_t)b2*1056 + 1032)*64 + (size_t)j*4;
    *(v4h*)(Qb+off) = z; *(v4h*)(Kb+off) = z;
  }
  { // Vt: 128*64 rows of 1056 tokens; pad = 24 halfs = 6 v4h per row
    int row = i / 6, j = i - row*6;
    size_t off = (size_t)row*1056 + 1032 + (size_t)j*4;
    *(v4h*)(Vt+off) = z;
  }
}

// ---------------- pass 1: QKV = x @ W^T + b; 32x32 tile per wave (2x2 WMMA) ----------------
__global__ __launch_bounds__(256) void qkv_gemm(const u16* __restrict__ xb, const u16* __restrict__ wb,
                                                const float* __restrict__ bias,
                                                u16* __restrict__ Qb, u16* __restrict__ Kb, u16* __restrict__ Vt){
  int wv = threadIdx.x >> 5, lane = threadIdx.x & 31;
  int n = lane & 15, half = lane >> 4;
  int wid = blockIdx.x*8 + wv;                        // 258*96 = 24768 tiles of 32x32
  int tileN = wid % 96, tileM = wid / 96;
  const u16* arow0 = xb + (size_t)(tileM*32 + n)*1024;
  const u16* arow1 = arow0 + (size_t)16*1024;
  const u16* brow0 = wb + (size_t)(tileN*32 + n)*1024;
  const u16* brow1 = brow0 + (size_t)16*1024;
  v8f c00 = {}, c01 = {}, c10 = {}, c11 = {};
  #pragma unroll 2
  for (int k = 0; k < 1024; k += 32){
    v16bf a0 = ldA(arow0, k, half), a1 = ldA(arow1, k, half);
    v16bf b0 = ldB(brow0, k, half), b1 = ldB(brow1, k, half);
    c00 = wmma_bf16(a0, b0, c00);
    c01 = wmma_bf16(a0, b1, c01);
    c10 = wmma_bf16(a1, b0, c10);
    c11 = wmma_bf16(a1, b1, c11);
  }
  #pragma unroll
  for (int mt = 0; mt < 2; ++mt){
    int m0 = tileM*32 + mt*16 + half*8;
    int b0 = m0 / 1032, t0 = m0 - b0*1032;            // one divide per subtile row-group
    #pragma unroll
    for (int nt = 0; nt < 2; ++nt){
      v8f cc = mt ? (nt ? c11 : c10) : (nt ? c01 : c00);
      int col = tileN*32 + nt*16 + n;
      float bv = bias[col];
      int which = col >> 10, hh = (col >> 6) & 15, d = col & 63;
      if (which == 2){
        #pragma unroll
        for (int r = 0; r < 8; ++r){
          int t = t0 + r, bb = b0;
          if (t >= 1032){ t -= 1032; ++bb; }
          Vt[(((size_t)hh*8 + bb)*64 + d)*1056 + t] = f2bf(cc[r] + bv);
        }
      } else {
        u16* dst = which ? Kb : Qb;
        #pragma unroll
        for (int r = 0; r < 8; ++r){
          int t = t0 + r, bb = b0;
          if (t >= 1032){ t -= 1032; ++bb; }
          dst[(((size_t)hh*8 + bb)*1056 + t)*64 + d] = f2bf(cc[r] + bv);
        }
      }
    }
  }
}

// ---------------- pass 2: factored rel-pos bias rows ----------------
__global__ void relpos_kernel(const u16* __restrict__ Qb, const float* __restrict__ rph,
                              const float* __restrict__ rpw,
                              float* __restrict__ relh, float* __restrict__ relw){
  int tok = blockIdx.x*4 + (threadIdx.x >> 6);        // 128*1024 tokens
  int sub = threadIdx.x & 63;
  int b2 = tok >> 10, q = tok & 1023;
  int qh = q >> 5, qw = q & 31;
  const v4h* q4 = (const v4h*)(Qb + ((size_t)b2*1056 + q)*64);   // UNscaled q row
  int kk = sub & 31;
  const v4f* r4 = (const v4f*)((sub < 32) ? (rph + (qh - kk + 31)*64)
                                          : (rpw + (qw - kk + 31)*64));
  float acc = 0.f;
  #pragma unroll
  for (int i = 0; i < 16; ++i){
    v4h qv = q4[i]; v4f rv = r4[i];
    acc += bf2f(qv[0])*rv[0] + bf2f(qv[1])*rv[1] + bf2f(qv[2])*rv[2] + bf2f(qv[3])*rv[3];
  }
  float* outp = (sub < 32) ? relh : relw;
  outp[((size_t)b2*1024 + q)*32 + kk] = acc;
}

// ---------------- pass 3: flash attention ----------------
#define ATTN_WAVES 4
__global__ __launch_bounds__(128) void attn_kernel(const u16* __restrict__ Qb, const u16* __restrict__ Kb,
                                                   const u16* __restrict__ Vt,
                                                   const float* __restrict__ relh, const float* __restrict__ relw,
                                                   u16* __restrict__ Ob){
  __shared__ __align__(16) float biasH[ATTN_WAVES][16][32];
  __shared__ __align__(16) u16   Pl[ATTN_WAVES][16][32];
  int w = threadIdx.x >> 5, lane = threadIdx.x & 31;
  int n = lane & 15, half = lane >> 4;
  int tile = blockIdx.x*ATTN_WAVES + w;
  if (tile >= 65) return;                             // tiles cover rows < 1040
  int b2 = blockIdx.y;
  int qrow0 = tile*16;

  // stage biasH rows (zero for non-grid q rows); biasW terms go to registers
  for (int r = 0; r < 16; ++r){
    int qrow = qrow0 + r;
    biasH[w][r][lane] = (qrow < 1024) ? relh[((size_t)b2*1024 + qrow)*32 + lane] : 0.f;
  }
  float bw0[8], bw1[8];
  #pragma unroll
  for (int r = 0; r < 8; ++r){
    int qrow = qrow0 + half*8 + r;
    bw0[r] = 0.f; bw1[r] = 0.f;
    if (qrow < 1024){
      const float* rw = relw + ((size_t)b2*1024 + qrow)*32;
      bw0[r] = rw[n]; bw1[r] = rw[16 + n];
    }
  }
  const u16* qrp = Qb + ((size_t)b2*1056 + qrow0 + n)*64;
  v16bf a0 = ldA(qrp, 0,  half);
  v16bf a1 = ldA(qrp, 32, half);
  const u16* krbase = Kb + ((size_t)b2*1056 + n)*64;
  const u16* vcbase = Vt + ((size_t)b2*64 + n)*1056;  // +cb*16*1056 per 16-col block
  v8f oacc[4]; float mrow[8], lrow[8];
  #pragma unroll
  for (int cb=0; cb<4; ++cb) oacc[cb] = (v8f){};
  #pragma unroll
  for (int r=0; r<8; ++r){ mrow[r] = -1e30f; lrow[r] = 0.f; }
  __builtin_amdgcn_wave_barrier();

  for (int jb = 0; jb < 33; ++jb){                    // 33 * 32 = 1056 keys
    int kk0 = jb*32;
    // prefetch next key block (global_prefetch_b8; no counter cost)
    if (jb < 32){
      int kn = kk0 + 32;
      __builtin_prefetch(krbase + (size_t)kn*64, 0, 3);
      __builtin_prefetch(krbase + (size_t)(kn+16)*64, 0, 3);
      __builtin_prefetch(vcbase + kn, 0, 3);
      __builtin_prefetch(vcbase + (size_t)16*1056 + kn, 0, 3);
      __builtin_prefetch(vcbase + (size_t)32*1056 + kn, 0, 3);
      __builtin_prefetch(vcbase + (size_t)48*1056 + kn, 0, 3);
    }
    const u16* kr0 = krbase + (size_t)kk0*64;
    const u16* kr1 = kr0 + 16*64;
    v8f cs0 = {}, cs1 = {};
    cs0 = wmma_bf16(a0, ldB(kr0, 0, half),  cs0);
    cs0 = wmma_bf16(a1, ldB(kr0, 32, half), cs0);
    cs1 = wmma_bf16(a0, ldB(kr1, 0, half),  cs1);
    cs1 = wmma_bf16(a1, ldB(kr1, 32, half), cs1);

    bool useb  = (jb < 32);                           // k0>>5 == k1>>5 == jb, uniform
    bool mask1 = (jb == 32);                          // keys 1040..1055 all pad
    bool mask0 = mask1 && (n >= 8);                   // keys 1024..1031 valid registers
    float s0[8], s1[8];
    #pragma unroll
    for (int r=0;r<8;++r){
      int m = half*8 + r;
      float t0 = cs0[r]*0.125f, t1 = cs1[r]*0.125f;   // scale = 64^-0.5
      if (useb){
        float bh = biasH[w][m][jb];
        t0 += bh + bw0[r];
        t1 += bh + bw1[r];
      }
      if (mask0) t0 = -1e30f;
      if (mask1) t1 = -1e30f;
      s0[r]=t0; s1[r]=t1;
    }
    // online softmax update (row = 16 lanes of one half-wave)
    #pragma unroll
    for (int r=0;r<8;++r){
      float mx = fmaxf(s0[r], s1[r]);
      mx = fmaxf(mx, __shfl_xor(mx, 1, 16));
      mx = fmaxf(mx, __shfl_xor(mx, 2, 16));
      mx = fmaxf(mx, __shfl_xor(mx, 4, 16));
      mx = fmaxf(mx, __shfl_xor(mx, 8, 16));
      float newm = fmaxf(mrow[r], mx);
      float alpha = __expf(mrow[r] - newm);
      float p0 = __expf(s0[r] - newm);
      float p1 = __expf(s1[r] - newm);
      float rs = p0 + p1;
      rs += __shfl_xor(rs, 1, 16);
      rs += __shfl_xor(rs, 2, 16);
      rs += __shfl_xor(rs, 4, 16);
      rs += __shfl_xor(rs, 8, 16);
      lrow[r] = lrow[r]*alpha + rs;
      mrow[r] = newm;
      oacc[0][r]*=alpha; oacc[1][r]*=alpha; oacc[2][r]*=alpha; oacc[3][r]*=alpha;
      int m = half*8 + r;
      Pl[w][m][n]    = f2bf(p0);
      Pl[w][m][16+n] = f2bf(p1);
    }
    __builtin_amdgcn_wave_barrier();

    // O += P @ V : A-frag of P from LDS, B-frags straight from transposed V (contiguous)
    v16bf ap = ldA(&Pl[w][n][0], 0, half);
    #pragma unroll
    for (int cb=0; cb<4; ++cb){
      const u16* vcol = vcbase + (size_t)cb*16*1056 + kk0;
      oacc[cb] = wmma_bf16(ap, ldB(vcol, 0, half), oacc[cb]);
    }
    __builtin_amdgcn_wave_barrier();
  }
  float rinv[8];
  #pragma unroll
  for (int r=0;r<8;++r) rinv[r] = 1.f / lrow[r];
  #pragma unroll
  for (int cb=0;cb<4;++cb){
    #pragma unroll
    for (int r=0;r<8;++r){
      int qrow = qrow0 + half*8 + r;
      if (qrow < 1032)
        Ob[((size_t)b2*1056 + qrow)*64 + cb*16 + n] = f2bf(oacc[cb][r]*rinv[r]);
    }
  }
}

// ---------------- pass 4: out = O_cat @ proj_w^T + b; 32x32 tile per wave ----------------
__global__ __launch_bounds__(256) void proj_gemm(const u16* __restrict__ Ob, const u16* __restrict__ wb,
                                                 const float* __restrict__ bias, float* __restrict__ out){
  int wv = threadIdx.x >> 5, lane = threadIdx.x & 31;
  int n = lane & 15, half = lane >> 4;
  int wid = blockIdx.x*8 + wv;                        // 258*32 = 8256 tiles of 32x32
  int tileN = wid & 31, tileM = wid >> 5;
  // per-mt row bases into head-major O: addr = rowbase + head*540672 + d
  size_t rowbase[2];
  #pragma unroll
  for (int mt = 0; mt < 2; ++mt){
    int m = tileM*32 + mt*16 + n;
    int bb = m / 1032, t = m - bb*1032;
    rowbase[mt] = ((size_t)bb*1056 + t)*64;
  }
  const u16* brow0 = wb + (size_t)(tileN*32 + n)*1024;
  const u16* brow1 = brow0 + (size_t)16*1024;
  v8f c00 = {}, c01 = {}, c10 = {}, c11 = {};
  for (int k = 0; k < 1024; k += 32){
    int k0 = k + half*8, k1 = k + 16 + half*8;        // 8-element runs stay in one head
    size_t o0 = (size_t)(k0 >> 6)*540672 + (k0 & 63); // 540672 = 8*1056*64
    size_t o1 = (size_t)(k1 >> 6)*540672 + (k1 & 63);
    BF16x16 fa0, fa1;
    fa0.q[0] = *(const v4u*)(Ob + rowbase[0] + o0);
    fa0.q[1] = *(const v4u*)(Ob + rowbase[0] + o1);
    fa1.q[0] = *(const v4u*)(Ob + rowbase[1] + o0);
    fa1.q[1] = *(const v4u*)(Ob + rowbase[1] + o1);
    v16bf b0 = ldB(brow0, k, half), b1 = ldB(brow1, k, half);
    c00 = wmma_bf16(fa0.v, b0, c00);
    c01 = wmma_bf16(fa0.v, b1, c01);
    c10 = wmma_bf16(fa1.v, b0, c10);
    c11 = wmma_bf16(fa1.v, b1, c11);
  }
  #pragma unroll
  for (int mt = 0; mt < 2; ++mt){
    int m0 = tileM*32 + mt*16 + half*8;
    #pragma unroll
    for (int nt = 0; nt < 2; ++nt){
      v8f cc = mt ? (nt ? c11 : c10) : (nt ? c01 : c00);
      int col = tileN*32 + nt*16 + n;
      float bv = bias[col];
      #pragma unroll
      for (int r = 0; r < 8; ++r)
        out[(size_t)(m0 + r)*1024 + col] = cc[r] + bv;
    }
  }
}

extern "C" void kernel_launch(void* const* d_in, const int* in_sizes, int n_in,
                              void* d_out, int out_size, void* d_ws, size_t ws_size,
                              hipStream_t stream){
  (void)in_sizes; (void)n_in; (void)out_size; (void)ws_size;
  const float* x     = (const float*)d_in[0];   // (8,1032,1024)
  const float* qkvw  = (const float*)d_in[1];   // (3072,1024)
  const float* qkvb  = (const float*)d_in[2];   // (3072)
  const float* projw = (const float*)d_in[3];   // (1024,1024)
  const float* projb = (const float*)d_in[4];   // (1024)
  const float* rph   = (const float*)d_in[5];   // (63,64)
  const float* rpw   = (const float*)d_in[6];   // (63,64)
  float* out = (float*)d_out;

  // workspace layout (~128 MB)
  char* p = (char*)d_ws;
  const size_t szQ = (size_t)128*1056*64;       // padded tokens, head-major
  u16* Qb = (u16*)p; p += szQ*2;
  u16* Kb = (u16*)p; p += szQ*2;
  u16* Vt = (u16*)p; p += szQ*2;                // transposed: [b2][d][token]
  u16* Ob = (u16*)p; p += szQ*2;
  float* relh = (float*)p; p += (size_t)128*1024*32*4;
  float* relw = (float*)p; p += (size_t)128*1024*32*4;
  u16* xb  = (u16*)p; p += (size_t)8256*1024*2;
  u16* qwb = (u16*)p; p += (size_t)3072*1024*2;
  u16* pwb = (u16*)p; p += (size_t)1024*1024*2;

  cvt_kernel<<<8256, 256, 0, stream>>>(x,     xb,  8256*1024/4);
  cvt_kernel<<<3072, 256, 0, stream>>>(qkvw,  qwb, 3072*1024/4);
  cvt_kernel<<<1024, 256, 0, stream>>>(projw, pwb, 1024*1024/4);
  pad_kernel<<<192, 256, 0, stream>>>(Qb, Kb, Vt);
  qkv_gemm<<<3096, 256, 0, stream>>>(xb, qwb, qkvb, Qb, Kb, Vt);
  relpos_kernel<<<32768, 256, 0, stream>>>(Qb, rph, rpw, relh, relw);
  attn_kernel<<<dim3(17,128), 128, 0, stream>>>(Qb, Kb, Vt, relh, relw, Ob);
  proj_gemm<<<1032, 256, 0, stream>>>(Ob, pwb, projb, out);
}
